// TwoLayerTransformer_39101382262894
// MI455X (gfx1250) — matrix operand order repaired and checked by reading, not compiled, not run on hardware
//
#include <hip/hip_runtime.h>
#include <hip/hip_bf16.h>
#include <math.h>

// Problem constants (from reference setup_inputs)
constexpr int T_  = 2048;
constexpr int H_  = 4;
constexpr int BS_ = 32;
constexpr int D_  = 128;
constexpr float EPSV = 1e-7f;

constexpr int ERLEN = T_ + 160;     // reversed exp(W) table, zero padded (out-of-band -> 0)
constexpr int KC    = 32;           // K chunk (WMMA K)
constexpr int XTPAD = 2;
constexpr int XTS   = KC + XTPAD;   // 34 halfs row stride in LDS (odd dwords -> no bank conflicts)

// Workspace layout (float offsets)
constexpr int OFF_E   = 0;                       // E[h][k]           H*T
constexpr int OFF_S   = OFF_E  + H_ * T_;        // S[h][t] prefix    H*T
constexpr int OFF_UT  = OFF_S  + H_ * T_;        // uT[b][h][d]       BS*H*D
constexpr int OFF_DT  = OFF_UT + BS_ * H_ * D_;  // denomT[b][h]      BS*H
constexpr int OFF_N2  = OFF_DT + BS_ * H_;       // ||u_t||^2         BS*H*T
constexpr int OFF_PT  = OFF_N2 + BS_ * H_ * T_;  // <u_t,u_T>         BS*H*T
constexpr int OFF_K   = OFF_PT + BS_ * H_ * T_;  // kernel[b][t]      BS*T
constexpr int OFF_END = OFF_K  + BS_ * T_;       // then f16 Erev[h][ERLEN]

typedef __attribute__((ext_vector_type(16))) _Float16 v16h;
typedef __attribute__((ext_vector_type(8)))  float    v8f;

// ---------------- K0: E = exp(W), reversed padded f16 table, prefix sums ----------------
__global__ __launch_bounds__(256) void k0_prep(const float* __restrict__ W,
                                               float* __restrict__ E,
                                               float* __restrict__ S,
                                               _Float16* __restrict__ Erev) {
  const int tid = threadIdx.x;
  for (int i = tid; i < H_ * T_; i += 256) {
    int h = i / T_, k = i % T_;
    E[h * T_ + k] = expf(W[k * H_ + h]);
  }
  __syncthreads();
  for (int i = tid; i < H_ * ERLEN; i += 256) {
    int h = i / ERLEN, r = i % ERLEN;
    Erev[h * ERLEN + r] = (r < T_) ? (_Float16)E[h * T_ + (T_ - 1 - r)] : (_Float16)0.0f;
  }
  if (tid < H_) {
    float s = 0.0f;
    for (int t = 0; t < T_; ++t) { s += E[tid * T_ + t]; S[tid * T_ + t] = s; }
  }
}

// ---------------- K1: u_T matvec + its norm ----------------
__global__ __launch_bounds__(128) void k1_uT(const float* __restrict__ X,
                                             const float* __restrict__ E,
                                             const float* __restrict__ S,
                                             float* __restrict__ UT,
                                             float* __restrict__ denomT) {
  const int b = blockIdx.x / H_;
  const int h = blockIdx.x % H_;
  const int d = threadIdx.x;
  const float* Eh = E + h * T_;
  const float* Xb = X + (size_t)b * T_ * D_;
  float acc = 0.0f;
  for (int c = 1; c < T_; ++c) acc += Eh[T_ - c] * Xb[(size_t)c * D_ + d];
  UT[((size_t)(b * H_ + h)) * D_ + d] = acc;
  __shared__ float red[128];
  red[d] = acc * acc;
  __syncthreads();
  for (int s = 64; s > 0; s >>= 1) { if (d < s) red[d] += red[d + s]; __syncthreads(); }
  if (d == 0) denomT[b * H_ + h] = sqrtf(red[0]) + EPSV * S[h * T_ + (T_ - 1)];
}

// ---------------- K2: banded Toeplitz GEMM via WMMA, fused norm/dot epilogue ----------------
__global__ __launch_bounds__(256) void k2_bandmm(const float* __restrict__ X,
                                                 const _Float16* __restrict__ Erev,
                                                 const float* __restrict__ UT,
                                                 float* __restrict__ n2,
                                                 float* __restrict__ pt) {
  const int rt = blockIdx.x;      // row tile 0..15 (128 rows each)
  const int h  = blockIdx.y;      // head
  const int b  = blockIdx.z;      // batch
  const int tid = threadIdx.x;
  const int wave = tid >> 5;
  const int lane = tid & 31;
  const int n = lane & 15;        // M for A, N for B/C
  const int halfSel = lane >> 4;

  __shared__ _Float16 sE[ERLEN];         // reversed exp(W[.,h]) with zero pad
  __shared__ _Float16 sXT[D_ * XTS];     // transposed X chunk: [d][c]
  __shared__ float    sUT[D_];

  for (int i = tid; i < ERLEN; i += 256) sE[i] = Erev[h * ERLEN + i];
  if (tid < D_) sUT[tid] = UT[((size_t)(b * H_ + h)) * D_ + tid];

  const float* Xb = X + (size_t)b * T_ * D_;
  const int r0 = rt * 128 + wave * 16;   // this wave's 16 output rows
  const int cMax = (rt + 1) * 128;

  v8f acc[8];
  #pragma unroll
  for (int j = 0; j < 8; ++j)
    #pragma unroll
    for (int e = 0; e < 8; ++e) acc[j][e] = 0.0f;

  for (int c0 = 0; c0 < cMax; c0 += KC) {
    __syncthreads();
    // Stage X[c0:c0+32, 0:128] transposed into LDS as f16 (coalesced global reads)
    #pragma unroll
    for (int i = 0; i < 16; ++i) {
      int lin = i * 256 + tid;
      int cc = lin >> 7;            // 0..31
      int dd = lin & 127;           // 0..127
      sXT[dd * XTS + cc] = (_Float16)Xb[(size_t)(c0 + cc) * D_ + dd];
    }
    __syncthreads();
    if (c0 <= r0 + 15) {            // wave-uniform: band intersects this wave's rows
      // A fragment: A[M][k] = E[(r0+M) - (c0+k)]; lane holds M=n,
      // K = 8*halfSel + {0..7} (elems 0..7) and 16 + 8*halfSel + {0..7} (elems 8..15).
      const int s0 = (T_ - 1) - (r0 + n - c0) + 8 * halfSel;
      v16h af;
      #pragma unroll
      for (int e = 0; e < 8; ++e) { af[e] = sE[s0 + e]; af[8 + e] = sE[s0 + 16 + e]; }
      #pragma unroll
      for (int j = 0; j < 8; ++j) {
        // B fragment: lane holds N = n (col d = 16j+n), K = 8*halfSel + {0..7} (elems 0..7) and 16 + 8*halfSel + {0..7} (elems 8..15)
        const _Float16* p = &sXT[(16 * j + n) * XTS + 8 * halfSel];
        v16h bf;
        #pragma unroll
        for (int e = 0; e < 8; ++e) { bf[e] = p[e]; bf[8 + e] = p[16 + e]; }
        acc[j] = __builtin_amdgcn_wmma_f32_16x16x32_f16(
            false, af, false, bf, (short)0, acc[j], false, false);
      }
    }
  }

  // Epilogue: per-row sum(u^2) and sum(u*uT); row (r0 + v + 8*halfSel) lives in
  // acc[j][v] across lanes of one half (N = n + 16j).
  float uT[8];
  #pragma unroll
  for (int j = 0; j < 8; ++j) uT[j] = sUT[16 * j + n];
  #pragma unroll
  for (int v = 0; v < 8; ++v) {
    float s2 = 0.0f, sp = 0.0f;
    #pragma unroll
    for (int j = 0; j < 8; ++j) { float x = acc[j][v]; s2 += x * x; sp += x * uT[j]; }
    #pragma unroll
    for (int m = 1; m < 16; m <<= 1) {   // masks <16 never cross lane-halves
      s2 += __shfl_xor(s2, m, 32);
      sp += __shfl_xor(sp, m, 32);
    }
    if (n == 0) {
      int row = r0 + v + 8 * halfSel;
      n2[((size_t)(b * H_ + h)) * T_ + row] = s2;
      pt[((size_t)(b * H_ + h)) * T_ + row] = sp;
    }
  }
}

// ---------------- K3: G -> combinatorial subset polynomial ----------------
__global__ __launch_bounds__(256) void k3_poly(const float* __restrict__ n2,
                                               const float* __restrict__ pt,
                                               const float* __restrict__ S,
                                               const float* __restrict__ denomT,
                                               const float* __restrict__ C_alpha,
                                               float* __restrict__ kern) {
  __shared__ float sC[32];
  if (threadIdx.x < 32) { float c = C_alpha[threadIdx.x]; sC[threadIdx.x] = c * c; }
  __syncthreads();
  const int idx = blockIdx.x * 256 + threadIdx.x;     // b*T + t
  const int b = idx / T_, t = idx % T_;
  float G[5];
  #pragma unroll
  for (int h = 0; h < H_; ++h) {
    float dt = sqrtf(n2[((size_t)(b * H_ + h)) * T_ + t]) + EPSV * S[h * T_ + t];
    G[h] = pt[((size_t)(b * H_ + h)) * T_ + t] / (dt * denomT[b * H_ + h]);
  }
  G[4] = 0.0f;   // identity channel vs zero last token -> exactly 0
  float k = 0.0f;
  #pragma unroll
  for (int num = 1; num < 32; ++num) {
    float p = sC[num];
    #pragma unroll
    for (int kk = 0; kk < 5; ++kk)
      if ((num >> (4 - kk)) & 1) p *= G[kk];
    k += p;
  }
  kern[idx] = k;
}

// ---------------- K4: softmax over t + probs . X ----------------
__global__ __launch_bounds__(256) void k4_out(const float* __restrict__ X,
                                              const float* __restrict__ kern,
                                              const float* __restrict__ aPtr,
                                              float* __restrict__ y) {
  const int b = blockIdx.x;
  const int tid = threadIdx.x;
  __shared__ float w[T_];
  __shared__ float red[256];
  const float a = aPtr[0];
  float m = -3.4e38f;
  for (int t = tid; t < T_; t += 256) {
    float v = a * kern[b * T_ + t];
    w[t] = v;
    m = fmaxf(m, v);
  }
  red[tid] = m; __syncthreads();
  for (int s = 128; s > 0; s >>= 1) { if (tid < s) red[tid] = fmaxf(red[tid], red[tid + s]); __syncthreads(); }
  m = red[0]; __syncthreads();
  float z = 0.0f;
  for (int t = tid; t < T_; t += 256) { float e = expf(w[t] - m); w[t] = e; z += e; }
  red[tid] = z; __syncthreads();
  for (int s = 128; s > 0; s >>= 1) { if (tid < s) red[tid] += red[tid + s]; __syncthreads(); }
  const float invZ = 1.0f / red[0];
  __syncthreads();
  if (tid < D_) {
    const float* Xb = X + (size_t)b * T_ * D_;
    float acc = 0.0f;
    for (int t = 0; t < T_; ++t) acc += w[t] * Xb[(size_t)t * D_ + tid];
    y[b * D_ + tid] = acc * invZ;
  }
}

extern "C" void kernel_launch(void* const* d_in, const int* in_sizes, int n_in,
                              void* d_out, int out_size, void* d_ws, size_t ws_size,
                              hipStream_t stream) {
  (void)in_sizes; (void)n_in; (void)out_size; (void)ws_size;
  const float* X = (const float*)d_in[0];
  const float* W = (const float*)d_in[1];
  const float* a = (const float*)d_in[2];
  const float* C = (const float*)d_in[3];
  float* ws = (float*)d_ws;
  float* E      = ws + OFF_E;
  float* S      = ws + OFF_S;
  float* UT     = ws + OFF_UT;
  float* denomT = ws + OFF_DT;
  float* n2     = ws + OFF_N2;
  float* pt     = ws + OFF_PT;
  float* kern   = ws + OFF_K;
  _Float16* Erev = (_Float16*)(ws + OFF_END);   // ~2.5 MB total workspace

  k0_prep<<<1, 256, 0, stream>>>(W, E, S, Erev);
  k1_uT<<<BS_ * H_, 128, 0, stream>>>(X, E, S, UT, denomT);
  k2_bandmm<<<dim3(T_ / 128, H_, BS_), 256, 0, stream>>>(X, Erev, UT, n2, pt);
  k3_poly<<<(BS_ * T_) / 256, 256, 0, stream>>>(n2, pt, S, denomT, C, kern);
  k4_out<<<BS_, 256, 0, stream>>>(X, kern, a, (float*)d_out);
}
